// EdgeSAGELayer_8229157339893
// MI455X (gfx1250) — compile-verified, matrix operand order checked by reading, not run on hardware
//
#include <hip/hip_runtime.h>
#include <hip/hip_bf16.h>

typedef __attribute__((ext_vector_type(16))) __bf16 v16bf;
typedef __attribute__((ext_vector_type(8)))  float  v8f;
typedef int v4i __attribute__((vector_size(16)));
typedef v4i __attribute__((address_space(1))) gas_v4i;   // global
typedef v4i __attribute__((address_space(3))) las_v4i;   // LDS

#define N_NODES 100000
#define N_EDGES 1000000
#define EDGE_F  64
#define IN_C    128
#define OUT_C   256
#define K_TOT   192
#define KBLKS   6                 // 192 / 32
#define M_TILES (N_NODES / 16)    // 6250, exact
#define N_TILES (OUT_C / 16)      // 16
#define W_ELEMS (N_TILES * KBLKS * 32 * 16)   // 49152 bf16 per (hi|lo) region
#define HALF_BYTES 49152          // bytes of one packed region half (8 N-tiles)
#define HALF_ELEMS 24576          // bf16 elements per half

#if defined(__has_builtin)
#if __has_builtin(__builtin_amdgcn_global_load_async_to_lds_b128)
#define HAVE_ASYNC_LDS 1
#endif
#endif

// ---------- bf16 split helpers (bit-ops only; no cvt dependence) ----------
__device__ __forceinline__ __bf16 trunc_bf16(float v) {
    unsigned u = __builtin_bit_cast(unsigned, v);
    return __builtin_bit_cast(__bf16, (unsigned short)(u >> 16));
}
__device__ __forceinline__ float bf16_to_f(__bf16 h) {
    unsigned u = ((unsigned)__builtin_bit_cast(unsigned short, h)) << 16;
    return __builtin_bit_cast(float, u);
}

// ---------- kernel 1: zero the accumulator region ----------
__global__ void zero_kernel(float* buf, int n) {
    int i = blockIdx.x * blockDim.x + threadIdx.x;
    if (i < n) buf[i] = 0.0f;
}

// ---------- kernel 2: pack W into wave32 WMMA B-operand layout (bf16 hi/lo) ----------
// index = ((nt*KBLKS + kb)*32 + lane)*16 + j
// lane 0-15: column n = nt*16 + lane,      K runs {kb*32+0..7, kb*32+16..23}
// lane 16-31: column n = nt*16 + lane-16,  K runs {kb*32+8..15, kb*32+24..31}
__global__ void packW_kernel(const float* __restrict__ W,
                             __bf16* __restrict__ Whi, __bf16* __restrict__ Wlo) {
    int t = blockIdx.x * blockDim.x + threadIdx.x;     // 16*6*32 = 3072
    if (t >= N_TILES * KBLKS * 32) return;
    int lane = t & 31;
    int rem  = t >> 5;
    int kb   = rem % KBLKS;
    int nt   = rem / KBLKS;
    int n    = nt * 16 + (lane & 15);
    int kl   = (lane >> 4) ? 8 : 0;
    int base = t * 16;
    #pragma unroll
    for (int j = 0; j < 16; ++j) {
        int k = kb * 32 + kl + ((j < 8) ? j : (j - 8 + 16));
        float v = W[k * OUT_C + n];
        __bf16 h = trunc_bf16(v);
        Whi[base + j] = h;
        Wlo[base + j] = trunc_bf16(v - bf16_to_f(h));
    }
}

// ---------- kernel 3: per-edge weighting + scatter-add (one wave per edge) ----------
__global__ __launch_bounds__(256) void scatter_kernel(
        const int* __restrict__ edge_index, const float* __restrict__ edge_attr,
        const float* __restrict__ edge_weight, float* sums, float* cnt) {
    int lane = threadIdx.x & 31;
    int e = blockIdx.x * (blockDim.x >> 5) + (threadIdx.x >> 5);
    if (e >= N_EDGES) return;
    // speculative prefetch of the edge_attr stream a few KB ahead (dropped if OOB)
    __builtin_prefetch(edge_attr + (size_t)(e + 32) * EDGE_F + lane * 2, 0, 1);
    int tgt = edge_index[e];                    // row 0 of [2,E]
    float w = edge_weight[e];
    const float2 v = *(const float2*)(edge_attr + (size_t)e * EDGE_F + lane * 2);
    float* dst = sums + (size_t)tgt * EDGE_F + lane * 2;
    __hip_atomic_fetch_add(dst,     v.x * w, __ATOMIC_RELAXED, __HIP_MEMORY_SCOPE_AGENT);
    __hip_atomic_fetch_add(dst + 1, v.y * w, __ATOMIC_RELAXED, __HIP_MEMORY_SCOPE_AGENT);
    if (lane == 0)
        __hip_atomic_fetch_add(cnt + tgt, 1.0f, __ATOMIC_RELAXED, __HIP_MEMORY_SCOPE_AGENT);
}

// ---------- kernel 4: mean = sums / max(cnt,1) * node_weight  (in place) ----------
__global__ void mean_kernel(float* sums, const float* __restrict__ cnt,
                            const float* __restrict__ node_weight) {
    int i = blockIdx.x * blockDim.x + threadIdx.x;
    if (i >= N_NODES * EDGE_F) return;
    int node = i >> 6;
    float c = cnt[node];
    c = c > 1.0f ? c : 1.0f;
    sums[i] = (sums[i] / c) * node_weight[node];
}

// ---------- kernel 5: WMMA GEMM  out = relu([node_attr | mean] @ W + b) ----------
// One wave per 16-row M-tile (8 waves/block). W is staged half-at-a-time into
// LDS (98,304 B) via async global->LDS b128 copies (ASYNCcnt), then all waves
// read B operands from LDS (ds_load_b128). 3 wmma per 32-wide K block (bf16x3).
__global__ __launch_bounds__(256) void gemm_kernel(
        const float* __restrict__ node_attr, const float* __restrict__ mean,
        const __bf16* __restrict__ Whi, const __bf16* __restrict__ Wlo,
        const float* __restrict__ bias, float* __restrict__ out) {
    extern __shared__ char lds[];             // 98,304 B: [Whi half | Wlo half]

    const int tid    = threadIdx.x;
    const int lane   = tid & 31;
    const int waveId = blockIdx.x * (blockDim.x >> 5) + (tid >> 5);
    const bool active = (waveId < M_TILES);   // wave-uniform
    const int m0  = active ? waveId * 16 : 0;
    const int row = m0 + (lane & 15);
    const int kl  = (lane >> 4) ? 8 : 0;      // A-layout half-wave K shift

    // ---- load & split A (16 rows x 192 K) into registers, once ----
    v16bf a_hi[KBLKS], a_lo[KBLKS];
    if (active) {
        #pragma unroll
        for (int kb = 0; kb < KBLKS; ++kb) {
            const float* src;
            int base0;
            if (kb < 4) { src = node_attr + (size_t)row * IN_C;   base0 = kb * 32 + kl; }
            else        { src = mean      + (size_t)row * EDGE_F; base0 = kb * 32 - IN_C + kl; }
            const int base1 = base0 + 16;
            #pragma unroll
            for (int j = 0; j < 8; ++j) {
                float v0 = src[base0 + j];
                float v1 = src[base1 + j];
                __bf16 h0 = trunc_bf16(v0), h1 = trunc_bf16(v1);
                a_hi[kb][j]     = h0;  a_lo[kb][j]     = trunc_bf16(v0 - bf16_to_f(h0));
                a_hi[kb][j + 8] = h1;  a_lo[kb][j + 8] = trunc_bf16(v1 - bf16_to_f(h1));
            }
        }
    }

    const __bf16* sWhi = (const __bf16*)lds;
    const __bf16* sWlo = sWhi + HALF_ELEMS;

    #pragma unroll 1
    for (int h = 0; h < 2; ++h) {
        // ---- stage half h of packed W (hi + lo) into LDS ----
        const char* gsrc[2] = { (const char*)Whi + h * HALF_BYTES,
                                (const char*)Wlo + h * HALF_BYTES };
        #pragma unroll
        for (int r = 0; r < 2; ++r) {
            #pragma unroll
            for (int i = 0; i < HALF_BYTES / (256 * 16); ++i) {   // 12 iters
                int off = i * 4096 + tid * 16;
#ifdef HAVE_ASYNC_LDS
                __builtin_amdgcn_global_load_async_to_lds_b128(
                    (gas_v4i*)(gsrc[r] + off),
                    (las_v4i*)(lds + r * HALF_BYTES + off),
                    0, 0);
#else
                *(uint4*)(lds + r * HALF_BYTES + off) = *(const uint4*)(gsrc[r] + off);
#endif
            }
        }
#ifdef HAVE_ASYNC_LDS
#if __has_builtin(__builtin_amdgcn_s_wait_asynccnt)
        __builtin_amdgcn_s_wait_asynccnt(0);
#else
        asm volatile("s_wait_asynccnt 0" ::: "memory");
#endif
#endif
        __syncthreads();

        // ---- compute the 8 N-tiles covered by this half ----
        if (active) {
            #pragma unroll 1
            for (int ntl = 0; ntl < 8; ++ntl) {
                v8f acc = {};
                #pragma unroll
                for (int kb = 0; kb < KBLKS; ++kb) {
                    const int boff = ((ntl * KBLKS + kb) * 32 + lane) * 16;
                    v16bf b_hi = *(const v16bf*)(sWhi + boff);
                    v16bf b_lo = *(const v16bf*)(sWlo + boff);
                    acc = __builtin_amdgcn_wmma_f32_16x16x32_bf16(false, a_hi[kb], false, b_lo,
                                                                  (short)0, acc, false, false);
                    acc = __builtin_amdgcn_wmma_f32_16x16x32_bf16(false, a_lo[kb], false, b_hi,
                                                                  (short)0, acc, false, false);
                    acc = __builtin_amdgcn_wmma_f32_16x16x32_bf16(false, a_hi[kb], false, b_hi,
                                                                  (short)0, acc, false, false);
                }
                const int n  = (h * 8 + ntl) * 16 + (lane & 15);
                const float bv = bias[n];
                #pragma unroll
                for (int r = 0; r < 8; ++r) {
                    int m = m0 + r + ((lane >> 4) << 3);   // C/D: M = r + 8*(lane/16)
                    float v = acc[r] + bv;
                    out[(size_t)m * OUT_C + n] = v > 0.0f ? v : 0.0f;
                }
            }
        }
        __syncthreads();   // protect LDS before next stage overwrites it
    }
}

extern "C" void kernel_launch(void* const* d_in, const int* in_sizes, int n_in,
                              void* d_out, int out_size, void* d_ws, size_t ws_size,
                              hipStream_t stream) {
    const int*   edge_index  = (const int*)  d_in[0];   // [2, E] int32
    const float* edge_attr   = (const float*)d_in[1];   // [E, 64]
    const float* node_attr   = (const float*)d_in[2];   // [N, 128]
    const float* edge_weight = (const float*)d_in[3];   // [E]
    const float* node_weight = (const float*)d_in[4];   // [N]
    const float* W           = (const float*)d_in[5];   // [192, 256]
    const float* b           = (const float*)d_in[6];   // [256]
    float* out = (float*)d_out;                          // [N, 256]

    char* ws = (char*)d_ws;
    float*  sums = (float*)(ws);                                   // N*64 f32 = 25,600,000 B
    float*  cnt  = (float*)(ws + (size_t)N_NODES * EDGE_F * 4);    //  N  f32 =    400,000 B
    __bf16* Whi  = (__bf16*)(ws + 26000000);                       // 98,304 B (64B aligned)
    __bf16* Wlo  = (__bf16*)(ws + 26000000 + 98304);               // 98,304 B

    // 1) zero sums + cnt
    {
        int n = N_NODES * EDGE_F + N_NODES;
        zero_kernel<<<(n + 255) / 256, 256, 0, stream>>>(sums, n);
    }
    // 2) pack W into WMMA B layout (bf16 hi/lo)
    packW_kernel<<<(N_TILES * KBLKS * 32 + 255) / 256, 256, 0, stream>>>(W, Whi, Wlo);
    // 3) edge scatter: one wave per edge (8 waves / 256-thread block)
    scatter_kernel<<<(N_EDGES + 7) / 8, 256, 0, stream>>>(edge_index, edge_attr,
                                                          edge_weight, sums, cnt);
    // 4) finalize mean in place
    mean_kernel<<<(N_NODES * EDGE_F + 255) / 256, 256, 0, stream>>>(sums, cnt, node_weight);
    // 5) fused concat-GEMM + bias + ReLU via bf16x3 WMMA, W staged in LDS
    gemm_kernel<<<(M_TILES + 7) / 8, 256, 98304, stream>>>(node_attr, sums, Whi, Wlo, b, out);
}
